// Voxelizer_16346645528627
// MI455X (gfx1250) — matrix operand order repaired and checked by reading, not compiled, not run on hardware
//
#include <hip/hip_runtime.h>
#include <hip/hip_bf16.h>

// ---------------------------------------------------------------------------
// Voxelizer: vol[c,z,x,y] = sum_g zr[z,g]*xr[x,g]*yr[y,g]*dens[c,g]
// Fold dens into zr -> S[cz,g]; per cz-slice compute a 128x128x4096 GEMM
//   vol_cz = (Xr .* S_cz) * Yr^T   via v_wmma_f32_16x16x32_f16 (f32 accum).
// Operands (Xr/Yr/S, ~3MB f16) are precomputed into d_ws; they fit in L2/WGP$,
// so kernel 2 is matrix-core bound; HBM only sees the 8MB complex output.
// Each wave owns a 32x32 output tile (2 A-frags x 2 B-frags -> 4 WMMA per
// K-step against 10 b128 loads) to keep VMEM issue rate below matrix rate.
// ---------------------------------------------------------------------------

typedef _Float16 v8h  __attribute__((ext_vector_type(8)));
typedef _Float16 v16h __attribute__((ext_vector_type(16)));
typedef float    v8f  __attribute__((ext_vector_type(8)));

#define GNUM 4096
#define NZ   64
#define NX   128
#define NY   128

// ---------------- Phase 1: build f16 response matrices --------------------
// rows 0..127   -> Xr[x][g]  = exp(-0.5*prec_x[g]*(xc-px)^2)
// rows 128..255 -> Yr[y][g]  = exp(-0.5*prec_y[g]*(yc-py)^2)
// rows 256..319 -> S[c][z][g]= exp(-0.5*prec_z[g]*(zc-pz)^2) * dens[c][g]
__global__ __launch_bounds__(256)
void vox_responses(const float* __restrict__ pos,   // (G,3)
                   const float* __restrict__ cov,   // (G,3,3) diagonal precisions
                   const float* __restrict__ dr,    // (G,)
                   const float* __restrict__ di,    // (G,)
                   _Float16* __restrict__ Xr,
                   _Float16* __restrict__ Yr,
                   _Float16* __restrict__ S)
{
    int idx = blockIdx.x * blockDim.x + threadIdx.x;
    if (idx >= (NX + NY + NZ) * GNUM) return;
    int row = idx >> 12;          // / 4096
    int g   = idx & (GNUM - 1);

    if (row < NX) {
        float c = -1.0f + 2.0f * row / (float)(NX - 1);
        float d = c - pos[g * 3 + 1];
        Xr[(size_t)row * GNUM + g] = (_Float16)expf(-0.5f * cov[g * 9 + 4] * d * d);
    } else if (row < NX + NY) {
        int y = row - NX;
        float c = -1.0f + 2.0f * y / (float)(NY - 1);
        float d = c - pos[g * 3 + 2];
        Yr[(size_t)y * GNUM + g] = (_Float16)expf(-0.5f * cov[g * 9 + 8] * d * d);
    } else {
        int z = row - (NX + NY);
        float c = -1.0f + 2.0f * z / (float)(NZ - 1);
        float d = c - pos[g * 3 + 0];
        float zr = expf(-0.5f * cov[g * 9 + 0] * d * d);
        S[(size_t)z * GNUM + g]        = (_Float16)(zr * dr[g]);
        S[(size_t)(NZ + z) * GNUM + g] = (_Float16)(zr * di[g]);
    }
}

// ---------------- Phase 2: WMMA GEMM per cz slice --------------------------
// grid.x = NX/64 (x tiles), grid.y = 2*NZ (cz slices). 256 threads = 8 waves.
// Wave (xsub = w&1, ysub = w>>1) computes a 32(x) x 32(y) tile: 4 accums.
__global__ __launch_bounds__(256)
void vox_gemm(const _Float16* __restrict__ Xr,
              const _Float16* __restrict__ Yr,
              const _Float16* __restrict__ S,
              float* __restrict__ out)
{
    const int cz   = blockIdx.y;          // 0..127 ; c = cz/64, z = cz%64
    const int c    = cz >> 6;
    const int z    = cz & 63;
    const int lane = threadIdx.x & 31;
    const int wave = threadIdx.x >> 5;    // 0..7
    const int xsub = wave & 1;            // 2 x-subtiles of 32 rows
    const int ysub = wave >> 1;           // 4 y-subtiles of 32 cols

    const int xrow0 = blockIdx.x * 64 + xsub * 32;  // first of 32 x rows
    const int y0    = ysub * 32;                    // two 16-col groups
    const int y1    = y0 + 16;

    const int grp  = lane >> 4;           // half-wave group
    const int mrow = lane & 15;

    // A fragment (16x32 f16, ISA 7.12.2): lane holds row (lane&15),
    //   halfs[0..7] = K[grp*8 .. +7], halfs[8..15] = K[16+grp*8 .. +7].
    const _Float16* arow0 = Xr + (size_t)(xrow0 + mrow) * GNUM;
    const _Float16* arow1 = arow0 + (size_t)16 * GNUM;
    const _Float16* srow  = S + (size_t)cz * GNUM;
    const int aoff0 = grp * 8;
    const int aoff1 = 16 + grp * 8;

    // B fragment (32x16 f16): lane holds column (lane&15),
    //   halfs[0..7] = K[grp*8 .. +7], halfs[8..15] = K[16+grp*8 .. +7].
    const _Float16* brow0 = Yr + (size_t)(y0 + mrow) * GNUM + grp * 8;
    const _Float16* brow1 = Yr + (size_t)(y1 + mrow) * GNUM + grp * 8;

    v8f acc00 = {}, acc01 = {}, acc10 = {}, acc11 = {};

#pragma unroll 2
    for (int kb = 0; kb < GNUM; kb += 32) {
        // Diagonal scale chunk (shared by both A fragments)
        v8h slo = *(const v8h*)(srow + kb + aoff0);
        v8h shi = *(const v8h*)(srow + kb + aoff1);

        // A operands: Xr chunks scaled by S_cz (folds dens*zr into A)
        v8h a0lo = *(const v8h*)(arow0 + kb + aoff0);
        v8h a0hi = *(const v8h*)(arow0 + kb + aoff1);
        v8h a1lo = *(const v8h*)(arow1 + kb + aoff0);
        v8h a1hi = *(const v8h*)(arow1 + kb + aoff1);
        a0lo *= slo; a0hi *= shi;          // -> v_pk_mul_f16
        a1lo *= slo; a1hi *= shi;
        v16h a0, a1;
#pragma unroll
        for (int i = 0; i < 8; ++i) {
            a0[i] = a0lo[i]; a0[8 + i] = a0hi[i];
            a1[i] = a1lo[i]; a1[8 + i] = a1hi[i];
        }

        v8h b0lo = *(const v8h*)(brow0 + kb);
        v8h b0hi = *(const v8h*)(brow0 + kb + 16);
        v8h b1lo = *(const v8h*)(brow1 + kb);
        v8h b1hi = *(const v8h*)(brow1 + kb + 16);
        v16h b0, b1;
#pragma unroll
        for (int i = 0; i < 8; ++i) {
            b0[i] = b0lo[i]; b0[8 + i] = b0hi[i];
            b1[i] = b1lo[i]; b1[8 + i] = b1hi[i];
        }

        acc00 = __builtin_amdgcn_wmma_f32_16x16x32_f16(
            false, a0, false, b0, (short)0, acc00, false, false);
        acc01 = __builtin_amdgcn_wmma_f32_16x16x32_f16(
            false, a0, false, b1, (short)0, acc01, false, false);
        acc10 = __builtin_amdgcn_wmma_f32_16x16x32_f16(
            false, a1, false, b0, (short)0, acc10, false, false);
        acc11 = __builtin_amdgcn_wmma_f32_16x16x32_f16(
            false, a1, false, b1, (short)0, acc11, false, false);
    }

    // C/D layout: lane L -> N = L&15 ; VGPR r -> M = (L>>4)*8 + r.
    // Output is complex64 (z,x,y): float index = 2*((z*NX+x)*NY+y) + c.
    const int ncol  = lane & 15;
    const int mbase = (lane >> 4) * 8;
#pragma unroll
    for (int r = 0; r < 8; ++r) {
        int xa = xrow0 + mbase + r;        // rows of A-frag 0
        int xb = xa + 16;                  // rows of A-frag 1
        size_t basea = ((size_t)z * NX + xa) * NY;
        size_t baseb = ((size_t)z * NX + xb) * NY;
        out[(basea + (y0 + ncol)) * 2 + c] = acc00[r];
        out[(basea + (y1 + ncol)) * 2 + c] = acc01[r];
        out[(baseb + (y0 + ncol)) * 2 + c] = acc10[r];
        out[(baseb + (y1 + ncol)) * 2 + c] = acc11[r];
    }
}

// ---------------------------------------------------------------------------
extern "C" void kernel_launch(void* const* d_in, const int* in_sizes, int n_in,
                              void* d_out, int out_size, void* d_ws, size_t ws_size,
                              hipStream_t stream) {
    const float* pos = (const float*)d_in[0];   // (G,3)
    const float* cov = (const float*)d_in[1];   // (G,3,3)
    const float* dr  = (const float*)d_in[2];   // (G,)
    const float* di  = (const float*)d_in[3];   // (G,)
    float* out = (float*)d_out;

    // workspace layout (f16): Xr | Yr | S  = 1MB + 1MB + 1MB
    _Float16* Xr = (_Float16*)d_ws;
    _Float16* Yr = Xr + (size_t)NX * GNUM;
    _Float16* S  = Yr + (size_t)NY * GNUM;

    int total = (NX + NY + NZ) * GNUM;          // 1,310,720 threads
    vox_responses<<<(total + 255) / 256, 256, 0, stream>>>(pos, cov, dr, di, Xr, Yr, S);

    dim3 grid(NX / 64, 2 * NZ);                 // 2 x 128 workgroups
    vox_gemm<<<grid, 256, 0, stream>>>(Xr, Yr, S, out);
}
